// MPNN_65859028517322
// MI455X (gfx1250) — matrix operand order repaired and checked by reading, not compiled, not run on hardware
//
#include <hip/hip_runtime.h>
#include <hip/hip_bf16.h>

#define NNODEc 30000
#define NGRAPHc 100
#define DEGc 24
#define NEDGEc (NNODEc * DEGc)
#define EPSf 1e-8f
#define CUTOFFf 5.0f

typedef __bf16 bf16_t;
typedef __attribute__((ext_vector_type(16))) __bf16 v16bf;
typedef __attribute__((ext_vector_type(8)))  __bf16 v8bf;
typedef __attribute__((ext_vector_type(8)))  float  v8f;

// ---------------- helpers ----------------
__device__ __forceinline__ bf16_t f2bf(float f) {
  unsigned u = __builtin_bit_cast(unsigned, f);
  unsigned r = u + 0x7FFFu + ((u >> 16) & 1u);
  unsigned short h = (unsigned short)(r >> 16);
  return __builtin_bit_cast(bf16_t, h);
}
__device__ __forceinline__ float bf2f(bf16_t b) {
  unsigned short h = __builtin_bit_cast(unsigned short, b);
  return __builtin_bit_cast(float, (unsigned)h << 16);
}
// fast silu: x * rcp(1+exp(-x)) -> v_exp_f32 + v_rcp_f32, no IEEE divide
__device__ __forceinline__ float siluf(float x) {
  return x * __builtin_amdgcn_rcpf(1.f + __expf(-x));
}
__device__ __forceinline__ float sincf_(float x) {
  float px = 3.14159265358979f * x;
  if (fabsf(px) < 1e-6f) return 1.f;
  return __sinf(px) / px;
}

// ---------------- WMMA GEMM: C[M,N] = act(A[M,K] @ Wt[N,K]^T + b) ----------------
// A: bf16 row-major (ld = lda), Wt: bf16 N-major (row n = K contiguous values),
// C: bf16 row-major (ld = ldc). One 16x16 tile per wave, 8 waves => 128 rows/block.
// Weights staged in LDS once per block; two n-tiles in flight per iteration so the
// two WMMAs have independent accumulators (hides D->D hazard NOPs).
template <int NK, int ACT>
__global__ __launch_bounds__(256)
void k_gemm(const bf16_t* __restrict__ A, int lda,
            const bf16_t* __restrict__ Wt, const float* __restrict__ bias,
            bf16_t* __restrict__ C, int ldc, int M, int N)
{
  extern __shared__ char smem_raw[];
  bf16_t* sW = (bf16_t*)smem_raw;
  const int Kp = NK * 32;

  // cooperative Wt -> LDS (global_load_b128 / ds_store_b128)
  {
    const int nvec = (N * Kp) >> 3;
    v8bf* d = (v8bf*)sW;
    const v8bf* s = (const v8bf*)Wt;
    for (int v = threadIdx.x; v < nvec; v += 256) d[v] = s[v];
  }
  __syncthreads();

  const int lane = threadIdx.x & 31;
  const int wv   = threadIdx.x >> 5;
  const int row0 = blockIdx.x * 128 + wv * 16;
  const int arow = row0 + (lane & 15);
  const int ah   = (lane >> 4) << 3;   // A half: K offset 0/8 (ISA 16-bit A layout)
  const int bh   = (lane >> 4) << 3;   // B half: K offset 0/8 (same layout as A)
  const bool wfull = (row0 + 16 <= M); // wave-uniform: no per-lane guards needed

  v16bf af[NK];
  if (wfull) {
    const bf16_t* pa = A + (size_t)arow * lda + ah;
#pragma unroll
    for (int kt = 0; kt < NK; ++kt) {
      v8bf lo = *(const v8bf*)(pa + kt * 32);
      v8bf hi = *(const v8bf*)(pa + kt * 32 + 16);
      v16bf f;
#pragma unroll
      for (int i = 0; i < 8; ++i) { f[i] = lo[i]; f[i + 8] = hi[i]; }
      af[kt] = f;
    }
  } else {
    const bool av = arow < M;
#pragma unroll
    for (int kt = 0; kt < NK; ++kt) {
      v16bf f;
      if (av) {
        const bf16_t* p = A + (size_t)arow * lda + kt * 32 + ah;
        v8bf lo = *(const v8bf*)p;
        v8bf hi = *(const v8bf*)(p + 16);
#pragma unroll
        for (int i = 0; i < 8; ++i) { f[i] = lo[i]; f[i + 8] = hi[i]; }
      } else {
#pragma unroll
        for (int i = 0; i < 16; ++i) f[i] = f2bf(0.f);
      }
      af[kt] = f;
    }
  }

  const int nnt = N >> 4;  // always even for our layer widths
  for (int nt = 0; nt < nnt; nt += 2) {
    const bf16_t* wp0 = sW + (size_t)(nt * 16 + (lane & 15)) * Kp + bh;
    const bf16_t* wp1 = wp0 + (size_t)16 * Kp;
    v8f acc0 = {}, acc1 = {};
#pragma unroll
    for (int kt = 0; kt < NK; ++kt) {
      v8bf l0 = *(const v8bf*)(wp0 + kt * 32);
      v8bf h0 = *(const v8bf*)(wp0 + kt * 32 + 16);
      v8bf l1 = *(const v8bf*)(wp1 + kt * 32);
      v8bf h1 = *(const v8bf*)(wp1 + kt * 32 + 16);
      v16bf b0, b1;
#pragma unroll
      for (int i = 0; i < 8; ++i) {
        b0[i] = l0[i]; b0[i + 8] = h0[i];
        b1[i] = l1[i]; b1[i + 8] = h1[i];
      }
      acc0 = __builtin_amdgcn_wmma_f32_16x16x32_bf16(false, af[kt], false, b0,
                                                     (short)0, acc0, false, false);
      acc1 = __builtin_amdgcn_wmma_f32_16x16x32_bf16(false, af[kt], false, b1,
                                                     (short)0, acc1, false, false);
    }
    const int col0 = nt * 16 + (lane & 15);
    const float bv0 = bias[col0];
    const float bv1 = bias[col0 + 16];
    const int rb = row0 + ((lane >> 4) << 3);
    if (wfull) {
#pragma unroll
      for (int r = 0; r < 8; ++r) {
        int row = rb + r;
        float x0 = acc0[r] + bv0;
        float x1 = acc1[r] + bv1;
        if (ACT) { x0 = siluf(x0); x1 = siluf(x1); }
        C[(size_t)row * ldc + col0] = f2bf(x0);
        C[(size_t)row * ldc + col0 + 16] = f2bf(x1);
      }
    } else {
#pragma unroll
      for (int r = 0; r < 8; ++r) {
        int row = rb + r;
        if (row < M) {
          float x0 = acc0[r] + bv0;
          float x1 = acc1[r] + bv1;
          if (ACT) { x0 = siluf(x0); x1 = siluf(x1); }
          C[(size_t)row * ldc + col0] = f2bf(x0);
          C[(size_t)row * ldc + col0 + 16] = f2bf(x1);
        }
      }
    }
  }
}

// ---------------- weight convert: W[K,N] f32 -> Wt[N,Kp] bf16 (zero-padded) -------
__global__ void k_conv_w(const float* __restrict__ W, bf16_t* __restrict__ Wt,
                         int K, int N, int Kp)
{
  int idx = blockIdx.x * blockDim.x + threadIdx.x;
  if (idx >= N * Kp) return;
  int n = idx / Kp, k = idx % Kp;
  float v = (k < K) ? W[(size_t)k * N + n] : 0.f;
  Wt[(size_t)n * Kp + k] = f2bf(v);
}

// ---------------- tiny species-pair MLP tables (16 rows) ----------------
__device__ void mlp_layer(const float* in, int ldi, const float* W, const float* B,
                          float* out, int ldo, int K, int N, int act)
{
  for (int idx = threadIdx.x; idx < 16 * N; idx += blockDim.x) {
    int r = idx / N, c = idx % N;
    float s = B[c];
    for (int k = 0; k < K; ++k) s += in[r * ldi + k] * W[k * N + c];
    if (act) s = siluf(s);
    out[r * ldo + c] = s;
  }
}

__global__ void k_pair_tables(const float* com_spec,
    const float* ncW0, const float* ncB0, const float* ncW1, const float* ncB1,
    const float* ncW2, const float* ncB2,
    const float* nnW0, const float* nnB0, const float* nnW1, const float* nnB1,
    const float* nnW2, const float* nnB2,
    const float* rwW0, const float* rwB0, const float* rwW1, const float* rwB1,
    const float* rwW2, const float* rwB2,
    float* emb_tab /*16x8*/, float* ead_tab /*16x40*/)
{
  __shared__ float h0[16 * 64], h1[16 * 64], ps[16 * 16];
  // neighcoeffnn: [8->64->64->16]
  mlp_layer(com_spec, 8, ncW0, ncB0, h0, 64, 8, 64, 1);  __syncthreads();
  mlp_layer(h0, 64, ncW1, ncB1, h1, 64, 64, 64, 1);      __syncthreads();
  mlp_layer(h1, 64, ncW2, ncB2, ps, 16, 64, 16, 0);      __syncthreads();
  // neighnn: [16->64->64->8]
  mlp_layer(ps, 16, nnW0, nnB0, h0, 64, 16, 64, 1);      __syncthreads();
  mlp_layer(h0, 64, nnW1, nnB1, h1, 64, 64, 64, 1);      __syncthreads();
  mlp_layer(h1, 64, nnW2, nnB2, emb_tab, 8, 64, 8, 0);   __syncthreads();
  // rweightnn: [16->64->64->40]
  mlp_layer(ps, 16, rwW0, rwB0, h0, 64, 16, 64, 1);      __syncthreads();
  mlp_layer(h0, 64, rwW1, rwB1, h1, 64, 64, 64, 1);      __syncthreads();
  mlp_layer(h1, 64, rwW2, rwB2, ead_tab, 40, 64, 40, 0);
}

// ---------------- geometry ----------------
__global__ void k_cell(const float* cell, const float* disp, float* cellp, float* symm)
{
  int idx = blockIdx.x * blockDim.x + threadIdx.x;
  if (idx >= NGRAPHc * 9) return;
  int g = idx / 9, r = (idx % 9) / 3, c = idx % 3;
  float sy = 0.5f * (disp[g * 9 + r * 3 + c] + disp[g * 9 + c * 3 + r]);
  symm[idx] = sy;
  float s = cell[idx];
  for (int k = 0; k < 3; ++k) {
    float sk = 0.5f * (disp[g * 9 + k * 3 + c] + disp[g * 9 + c * 3 + k]);
    s += cell[g * 9 + r * 3 + k] * sk;
  }
  cellp[idx] = s;
}

__global__ void k_cart(const float* cart, const float* symm, const int* clist,
                       const float* species, float* cartp, int* spec)
{
  int i = blockIdx.x * blockDim.x + threadIdx.x;
  if (i >= NNODEc) return;
  int g = clist[i];
  for (int k = 0; k < 3; ++k) {
    float s = cart[i * 3 + k];
    for (int j = 0; j < 3; ++j) s += cart[i * 3 + j] * symm[g * 9 + j * 3 + k];
    cartp[i * 3 + k] = s;
  }
  const float R[4] = {1.f, 6.f, 7.f, 8.f};
  float sp = species[i];
  int si = 0;
  for (int s = 0; s < 4; ++s) if (fabsf(sp - R[s]) < 0.5f) si = s;
  spec[i] = si;
}

__global__ void k_edge_geom(const float* cartp, const float* cellp, const int* nl,
                            const int* clist, const float* shiftim,
                            float* sph, float* cutf, float* ndist)
{
  int e = blockIdx.x * blockDim.x + threadIdx.x;
  if (e >= NEDGEc) return;
  int i0 = nl[e], i1 = nl[NEDGEc + e];
  int g = clist[i0];
  float s0 = shiftim[e], s1 = shiftim[NEDGEc + e], s2 = shiftim[2 * NEDGEc + e];
  float dv[3];
  for (int k = 0; k < 3; ++k) {
    float sh = s0 * cellp[g * 9 + k] + s1 * cellp[g * 9 + 3 + k] + s2 * cellp[g * 9 + 6 + k];
    dv[k] = cartp[i1 * 3 + k] - cartp[i0 * 3 + k] + sh;
  }
  float d2 = dv[0] * dv[0] + dv[1] * dv[1] + dv[2] * dv[2];
  float nf = d2 > EPSf ? 1.f : 0.f;
  float dist = sqrtf(d2 + EPSf);
  float x = dv[0] / dist, y = dv[1] / dist, z = dv[2] / dist;
  float s[9] = {1.f, x, y, z, x * y, y * z, 3.f * z * z - 1.f, x * z, x * x - y * y};
  float n0 = s[0] * s[0] + EPSf;
  float n1 = s[1] * s[1] + s[2] * s[2] + s[3] * s[3] + EPSf;
  float n2 = s[4] * s[4] + s[5] * s[5] + s[6] * s[6] + s[7] * s[7] + s[8] * s[8] + EPSf;
  float i0n = rsqrtf(n0);
  float i1n = rsqrtf(n1) * 1.73205080757f;  // sqrt(3)
  float i2n = rsqrtf(n2) * 2.2360679775f;   // sqrt(5)
  sph[(size_t)e * 9 + 0] = s[0] * i0n;
  for (int j = 1; j < 4; ++j) sph[(size_t)e * 9 + j] = s[j] * i1n;
  for (int j = 4; j < 9; ++j) sph[(size_t)e * 9 + j] = s[j] * i2n;
  float nd = dist / CUTOFFf;
  float nd2 = nd * nd, nd6 = nd2 * nd2 * nd2;
  float env = 1.f - nd6 * (28.f - 48.f * nd + 21.f * nd2);
  cutf[e] = env * env * nf;
  ndist[e] = nd;
}

__global__ void k_ave(const float* cutf, float* ave)
{
  int i = blockIdx.x * blockDim.x + threadIdx.x;
  if (i >= NNODEc) return;
  float s = EPSf;
  for (int e = i * DEGc; e < i * DEGc + DEGc; ++e) s += cutf[e];
  ave[i] = s;
}

// radial_func input (NEDGE x 32 bf16, cols 16..31 zero) + pair index
__global__ void k_radialin(const int* nl, const int* spec, const float* emb_tab,
                           const float* ead_tab, const float* cutf, const float* ndist,
                           bf16_t* A, int* pr)
{
  int e = blockIdx.x * blockDim.x + threadIdx.x;
  if (e >= NEDGEc) return;
  int p = spec[nl[e]] * 4 + spec[nl[NEDGEc + e]];
  pr[e] = p;
  float cut = cutf[e], nd = ndist[e];
  for (int c = 0; c < 8; ++c)
    A[(size_t)e * 32 + c] = f2bf(ead_tab[p * 40 + 32 + c] * cut);
  for (int c = 0; c < 8; ++c)
    A[(size_t)e * 32 + 8 + c] = f2bf(sincf_(nd * emb_tab[p * 8 + c]) * cut);
  for (int c = 16; c < 32; ++c) A[(size_t)e * 32 + c] = f2bf(0.f);
}

// ead init: cols 0..31 = smooth_ead[:, :32]; cols 32..63 = wradial[:, 3]
__global__ void k_eadinit(const int* pr, const float* ead_tab, const float* cutf,
                          const bf16_t* wrad, bf16_t* ead)
{
  long long idx = (long long)blockIdx.x * blockDim.x + threadIdx.x;
  if (idx >= (long long)NEDGEc * 64) return;
  int e = (int)(idx >> 6), c = (int)(idx & 63);
  if (c < 32) ead[(size_t)e * 128 + c] = f2bf(ead_tab[pr[e] * 40 + c] * cutf[e]);
  else        ead[(size_t)e * 128 + c] = wrad[(size_t)e * 128 + 96 + (c - 32)];
}

// density cols 0..31 = segsum(wradial[:, 2])
__global__ void k_dens0(const bf16_t* wrad, float* density)
{
  int idx = blockIdx.x * blockDim.x + threadIdx.x;
  if (idx >= NNODEc * 32) return;
  int i = idx >> 5, k = idx & 31;
  float s = 0.f;
  for (int e = i * DEGc; e < i * DEGc + DEGc; ++e)
    s += bf2f(wrad[(size_t)e * 128 + 64 + k]);
  density[(size_t)i * 128 + k] = s;
}

// out[i,j,k] = segsum_e( sph[e,j] * sum_t buf[e, oj + t*tstep + k] ) (/ave)
__global__ void k_edge_reduce(float* out, const bf16_t* buf, int ld,
                              int o0, int o1, int o2, int o3, int nt, int tstep,
                              const float* sph, const float* ave, int divave)
{
  int idx = blockIdx.x * blockDim.x + threadIdx.x;
  if (idx >= NNODEc * 128) return;
  int i = idx >> 7, jk = idx & 127, j = jk >> 5, k = jk & 31;
  int off = (j == 0 ? o0 : j == 1 ? o1 : j == 2 ? o2 : o3) + k;
  float s = 0.f;
  for (int e = i * DEGc; e < i * DEGc + DEGc; ++e) {
    float v = 0.f;
    for (int t = 0; t < nt; ++t) v += bf2f(buf[(size_t)e * ld + off + t * tstep]);
    s += v * sph[(size_t)e * 9 + j];
  }
  if (divave) s /= ave[i];
  out[idx] = s;
}

// out[i,j,m] (+)= scale * sum_k in[i,j,k] * tab[spec[i]*sistride + soff + k*32 + m] (/ave)
__global__ void k_contract(float* out, const float* in, const float* tab,
                           const int* spec, int sistride, int soff, float scale,
                           const float* ave, int divave, int accum)
{
  int idx = blockIdx.x * blockDim.x + threadIdx.x;
  if (idx >= NNODEc * 128) return;
  int i = idx >> 7, j = (idx >> 5) & 3, m = idx & 31;
  const float* T = tab + (size_t)spec[i] * sistride + soff;
  const float* I = in + (size_t)i * 128 + j * 32;
  float s = 0.f;
  for (int k = 0; k < 32; ++k) s += I[k] * T[k * 32 + m];
  s *= scale;
  if (divave) s /= ave[i];
  if (accum) out[idx] += s; else out[idx] = s;
}

__global__ void k_scalebuf(float* out, const float* in, float s, int n)
{
  int idx = blockIdx.x * blockDim.x + threadIdx.x;
  if (idx < n) out[idx] = in[idx] * s;
}

// norm_ead append into ead[:, dstcol..dstcol+31]
__global__ void k_normead(bf16_t* ead, int dstcol, const bf16_t* radial,
                          const float* ncorb, const float* sph, const int* nl)
{
  long long idx = (long long)blockIdx.x * blockDim.x + threadIdx.x;
  if (idx >= (long long)NEDGEc * 32) return;
  int e = (int)(idx >> 5), k = (int)(idx & 31);
  int i0 = nl[e], i1 = nl[NEDGEc + e];
  float s = 0.f;
  for (int j = 0; j < 4; ++j) {
    int pj = j ? 1 : 0;
    float r0 = bf2f(radial[(size_t)e * 192 + pj * 32 + k]);
    float r1 = bf2f(radial[(size_t)e * 192 + (2 + pj) * 32 + k]);
    float a = r0 * ncorb[(size_t)i0 * 128 + j * 32 + k]
            + r1 * ncorb[(size_t)i1 * 128 + j * 32 + k];
    s += sph[(size_t)e * 9 + j] * a;
  }
  ead[(size_t)e * 128 + dstcol + k] = f2bf(s * 0.70710678f);
}

// iter_orb[i,j,k] = segsum_e( corb[nl1[e],j,k] * orb[e, pj*32+k] ) / ave
__global__ void k_iterorb(float* out, const float* corb, const bf16_t* orb,
                          const int* nl, const float* ave)
{
  int idx = blockIdx.x * blockDim.x + threadIdx.x;
  if (idx >= NNODEc * 128) return;
  int i = idx >> 7, j = (idx >> 5) & 3, k = idx & 31;
  int pj = j ? 1 : 0;
  float s = 0.f;
  for (int e = i * DEGc; e < i * DEGc + DEGc; ++e) {
    int n1 = nl[NEDGEc + e];
    s += corb[(size_t)n1 * 128 + j * 32 + k] * bf2f(orb[(size_t)e * 160 + pj * 32 + k]);
  }
  out[idx] = s / ave[i];
}

__global__ void k_density1(float* density, int col, const float* sum_orb,
                           const float* ncorb)
{
  int idx = blockIdx.x * blockDim.x + threadIdx.x;
  if (idx >= NNODEc * 32) return;
  int i = idx >> 5, k = idx & 31;
  float s = 0.f;
  for (int j = 0; j < 4; ++j)
    s += sum_orb[(size_t)i * 128 + j * 32 + k] * ncorb[(size_t)i * 128 + j * 32 + k];
  density[(size_t)i * 128 + col + k] = s;
}

// center_orbital = tmpA * l_coeff[pj, spec, m] / sqrt(2)
__global__ void k_comb(float* co, const float* tmpA, const float* lc,
                       const int* spec)
{
  int idx = blockIdx.x * blockDim.x + threadIdx.x;
  if (idx >= NNODEc * 128) return;
  int i = idx >> 7, j = (idx >> 5) & 3, m = idx & 31;
  int pj = j ? 1 : 0;
  float f = lc[pj * 128 + spec[i] * 32 + m];
  co[idx] = tmpA[idx] * f * 0.70710678f;
}

__global__ void k_densin(const float* density, const float* ave, bf16_t* out)
{
  int idx = blockIdx.x * blockDim.x + threadIdx.x;
  if (idx >= NNODEc * 128) return;
  int i = idx >> 7;
  out[idx] = f2bf(density[idx] / ave[i]);
}

__global__ void k_atom(const bf16_t* hidden, const float* W2, const float* b2,
                       const float* scale, const int* spec, const float* cf,
                       float* atom)
{
  int i = blockIdx.x * blockDim.x + threadIdx.x;
  if (i >= NNODEc) return;
  float s = b2[0];
  for (int c = 0; c < 64; ++c) s += bf2f(hidden[(size_t)i * 64 + c]) * W2[c];
  int si = spec[i];
  atom[i] = (s * scale[si * 2] + scale[si * 2 + 1]) * cf[i];
}

__global__ void k_graph(const float* atom, float* out)
{
  __shared__ float sm[256];
  int g = blockIdx.x;
  float s = 0.f;
  for (int i = threadIdx.x; i < NNODEc / NGRAPHc; i += 256)
    s += atom[g * (NNODEc / NGRAPHc) + i];
  sm[threadIdx.x] = s;
  __syncthreads();
  for (int st = 128; st > 0; st >>= 1) {
    if ((int)threadIdx.x < st) sm[threadIdx.x] += sm[threadIdx.x + st];
    __syncthreads();
  }
  if (threadIdx.x == 0) out[1 + g] = sm[0];
}

__global__ void k_total(float* out)
{
  __shared__ float sm[128];
  float s = 0.f;
  for (int i = threadIdx.x; i < NGRAPHc; i += 128) s += out[1 + i];
  sm[threadIdx.x] = s;
  __syncthreads();
  for (int st = 64; st > 0; st >>= 1) {
    if ((int)threadIdx.x < st) sm[threadIdx.x] += sm[threadIdx.x + st];
    __syncthreads();
  }
  if (threadIdx.x == 0) out[0] = sm[0];
}

// ---------------- host ----------------
static inline void launch_gemm(hipStream_t st, const bf16_t* A, int lda,
                               const bf16_t* Wt, const float* b,
                               bf16_t* C, int ldc, int M, int N, int Kp, int act)
{
  int blocks = (M + 127) / 128;
  size_t sh = (size_t)N * Kp * 2;
#define LGEMM(NKv)                                                                  \
  do {                                                                              \
    if (act) k_gemm<NKv, 1><<<blocks, 256, sh, st>>>(A, lda, Wt, b, C, ldc, M, N);  \
    else     k_gemm<NKv, 0><<<blocks, 256, sh, st>>>(A, lda, Wt, b, C, ldc, M, N);  \
  } while (0)
  switch (Kp / 32) {
    case 1: LGEMM(1); break;
    case 2: LGEMM(2); break;
    case 3: LGEMM(3); break;
    case 4: LGEMM(4); break;
    default: LGEMM(5); break;
  }
#undef LGEMM
}

extern "C" void kernel_launch(void* const* d_in, const int* in_sizes, int n_in,
                              void* d_out, int out_size, void* d_ws, size_t ws_size,
                              hipStream_t stream)
{
  (void)in_sizes; (void)n_in; (void)out_size; (void)ws_size;
  // input map (dict insertion order, lists flattened depth-first, [W,b] pairs):
  const float* cart   = (const float*)d_in[0];
  const float* cell   = (const float*)d_in[1];
  const float* disp   = (const float*)d_in[2];
  const float* shiftim= (const float*)d_in[3];
  const float* cfac   = (const float*)d_in[4];
  const float* species= (const float*)d_in[5];
  const float* scale  = (const float*)d_in[6];
  const float* spec_co= (const float*)d_in[7];
  const float* con_co = (const float*)d_in[8];
  const float* l_co   = (const float*)d_in[9];
  // 10..15 neighcoeffnn, 16..21 neighnn, 22..27 rweightnn, 28..33 radialnn,
  // 34..45 mpnn(2x3), 46..63 ead(3x3), 64..67 outnn, 68 com_spec
  const float* com_spec = (const float*)d_in[68];
  const int* nl    = (const int*)d_in[69];
  const int* clist = (const int*)d_in[70];
  float* out = (float*)d_out;

  char* ws = (char*)d_ws;
  size_t off = 0;
  auto take = [&](size_t bytes) {
    size_t r = off;
    off += (bytes + 255) & ~(size_t)255;
    return r;
  };
  float* cellp  = (float*)(ws + take(NGRAPHc * 9 * 4));
  float* symm   = (float*)(ws + take(NGRAPHc * 9 * 4));
  float* cartp  = (float*)(ws + take((size_t)NNODEc * 3 * 4));
  int*   spec   = (int*)  (ws + take((size_t)NNODEc * 4));
  float* sph    = (float*)(ws + take((size_t)NEDGEc * 9 * 4));
  float* cutf   = (float*)(ws + take((size_t)NEDGEc * 4));
  float* ndist  = (float*)(ws + take((size_t)NEDGEc * 4));
  float* ave    = (float*)(ws + take((size_t)NNODEc * 4));
  int*   pr     = (int*)  (ws + take((size_t)NEDGEc * 4));
  float* emb_tab= (float*)(ws + take(16 * 8 * 4));
  float* ead_tab= (float*)(ws + take(16 * 40 * 4));
  char*  wpool  = ws + take(4u << 20);
  bf16_t* Ain   = (bf16_t*)(ws + take((size_t)NEDGEc * 32 * 2));
  bf16_t* act0  = (bf16_t*)(ws + take((size_t)NEDGEc * 128 * 2));
  bf16_t* act1  = (bf16_t*)(ws + take((size_t)NEDGEc * 128 * 2));
  bf16_t* eadb  = (bf16_t*)(ws + take((size_t)NEDGEc * 128 * 2));
  bf16_t* radb  = (bf16_t*)(ws + take((size_t)NEDGEc * 192 * 2));
  bf16_t* worb  = (bf16_t*)(ws + take((size_t)NEDGEc * 160 * 2)); // wradial (ld128) then orb_coeff (ld160)
  float* co     = (float*)(ws + take((size_t)NNODEc * 128 * 4));
  float* cotmp  = (float*)(ws + take((size_t)NNODEc * 128 * 4));
  float* ncorb  = (float*)(ws + take((size_t)NNODEc * 128 * 4));
  float* corb   = (float*)(ws + take((size_t)NNODEc * 128 * 4));
  float* iterb  = (float*)(ws + take((size_t)NNODEc * 128 * 4));
  float* sphb   = (float*)(ws + take((size_t)NNODEc * 128 * 4));
  float* tmpA   = (float*)(ws + take((size_t)NNODEc * 128 * 4));
  float* density= (float*)(ws + take((size_t)NNODEc * 128 * 4));
  bf16_t* densin= (bf16_t*)(ws + take((size_t)NNODEc * 128 * 2));
  bf16_t* hidden= (bf16_t*)(ws + take((size_t)NNODEc * 64 * 2));
  float* atom   = (float*)(ws + take((size_t)NNODEc * 4));

  // convert all GEMM weights to bf16 N-major
  size_t wo = 0;
  auto conv = [&](int widx, int K, int N, int Kp) {
    bf16_t* Wt = (bf16_t*)(wpool + wo);
    wo += (((size_t)N * Kp * 2) + 255) & ~(size_t)255;
    int tot = N * Kp;
    k_conv_w<<<(tot + 255) / 256, 256, 0, stream>>>((const float*)d_in[widx], Wt, K, N, Kp);
    return Wt;
  };
  bf16_t* rW0 = conv(28, 16, 128, 32);   const float* rB0 = (const float*)d_in[29];
  bf16_t* rW1 = conv(30, 128, 128, 128); const float* rB1 = (const float*)d_in[31];
  bf16_t* rW2 = conv(32, 128, 128, 128); const float* rB2 = (const float*)d_in[33];
  bf16_t* mW[2][3]; const float* mB[2][3];
  int mK0[2] = {96, 128};
  for (int it = 0; it < 2; ++it) {
    int base = 34 + it * 6;
    mW[it][0] = conv(base + 0, mK0[it], 128, mK0[it]); mB[it][0] = (const float*)d_in[base + 1];
    mW[it][1] = conv(base + 2, 128, 128, 128);         mB[it][1] = (const float*)d_in[base + 3];
    mW[it][2] = conv(base + 4, 128, 160, 128);         mB[it][2] = (const float*)d_in[base + 5];
  }
  bf16_t* eW[3][3]; const float* eB[3][3];
  int eK0[3] = {96, 128, 64};
  for (int m = 0; m < 3; ++m) {
    int base = 46 + m * 6;
    eW[m][0] = conv(base + 0, eK0[m], 128, eK0[m]); eB[m][0] = (const float*)d_in[base + 1];
    eW[m][1] = conv(base + 2, 128, 128, 128);       eB[m][1] = (const float*)d_in[base + 3];
    eW[m][2] = conv(base + 4, 128, 192, 128);       eB[m][2] = (const float*)d_in[base + 5];
  }
  bf16_t* oW0 = conv(64, 128, 64, 128); const float* oB0 = (const float*)d_in[65];
  const float* oW1 = (const float*)d_in[66];
  const float* oB1 = (const float*)d_in[67];

  // species-pair tables
  k_pair_tables<<<1, 256, 0, stream>>>(com_spec,
      (const float*)d_in[10], (const float*)d_in[11], (const float*)d_in[12],
      (const float*)d_in[13], (const float*)d_in[14], (const float*)d_in[15],
      (const float*)d_in[16], (const float*)d_in[17], (const float*)d_in[18],
      (const float*)d_in[19], (const float*)d_in[20], (const float*)d_in[21],
      (const float*)d_in[22], (const float*)d_in[23], (const float*)d_in[24],
      (const float*)d_in[25], (const float*)d_in[26], (const float*)d_in[27],
      emb_tab, ead_tab);

  // geometry
  k_cell<<<(NGRAPHc * 9 + 255) / 256, 256, 0, stream>>>(cell, disp, cellp, symm);
  k_cart<<<(NNODEc + 255) / 256, 256, 0, stream>>>(cart, symm, clist, species, cartp, spec);
  k_edge_geom<<<(NEDGEc + 255) / 256, 256, 0, stream>>>(cartp, cellp, nl, clist, shiftim,
                                                        sph, cutf, ndist);
  k_ave<<<(NNODEc + 255) / 256, 256, 0, stream>>>(cutf, ave);

  const int N128 = NNODEc * 128, N32 = NNODEc * 32;
  const long long E32 = (long long)NEDGEc * 32, E64 = (long long)NEDGEc * 64;

  // radialnn: [32p ->128 ->128 ->128], wradial lives in worb with ld 128
  k_radialin<<<(NEDGEc + 255) / 256, 256, 0, stream>>>(nl, spec, emb_tab, ead_tab, cutf,
                                                       ndist, Ain, pr);
  launch_gemm(stream, Ain, 32, rW0, rB0, act0, 128, NEDGEc, 128, 32, 1);
  launch_gemm(stream, act0, 128, rW1, rB1, act1, 128, NEDGEc, 128, 128, 1);
  launch_gemm(stream, act1, 128, rW2, rB2, worb, 128, NEDGEc, 128, 128, 0);

  k_eadinit<<<(int)((E64 + 255) / 256), 256, 0, stream>>>(pr, ead_tab, cutf, worb, eadb);
  k_dens0<<<(N32 + 255) / 256, 256, 0, stream>>>(worb, density);
  // center_orbital: reduce wradial[:, PINDEX_L]*sph, contract with spec_coeff/sqrt(32), /ave
  k_edge_reduce<<<(N128 + 255) / 256, 256, 0, stream>>>(cotmp, worb, 128,
      0, 32, 32, 32, 1, 0, sph, ave, 0);
  k_contract<<<(N128 + 255) / 256, 256, 0, stream>>>(co, cotmp, spec_co, spec,
      1024, 0, 0.17677669529f, ave, 1, 0);

  // pre-loop radial = ead[2](ead[64])
  launch_gemm(stream, eadb, 128, eW[2][0], eB[2][0], act0, 128, NEDGEc, 128, 64, 1);
  launch_gemm(stream, act0, 128, eW[2][1], eB[2][1], act1, 128, NEDGEc, 128, 128, 1);
  launch_gemm(stream, act1, 128, eW[2][2], eB[2][2], radb, 192, NEDGEc, 192, 128, 0);

  for (int it = 0; it < 2; ++it) {
    int eadK = 96 + 32 * it;
    // norm_corb = center_orbital / sqrt(2)
    k_scalebuf<<<(N128 + 255) / 256, 256, 0, stream>>>(ncorb, co, 0.70710678f, N128);
    // norm_ead append -> ead col 64+32*it
    k_normead<<<(int)((E32 + 255) / 256), 256, 0, stream>>>(eadb, 64 + 32 * it, radb,
                                                            ncorb, sph, nl);
    // sum_orb from radial[:,2,PINDEX_L]; density1 -> col 32+32*it
    k_edge_reduce<<<(N128 + 255) / 256, 256, 0, stream>>>(cotmp, radb, 192,
        128, 160, 160, 160, 1, 0, sph, ave, 0);
    k_density1<<<(N32 + 255) / 256, 256, 0, stream>>>(density, 32 + 32 * it, cotmp, ncorb);
    // orb_coeff = mpnn[it](ead)
    launch_gemm(stream, eadb, 128, mW[it][0], mB[it][0], act0, 128, NEDGEc, 128, eadK, 1);
    launch_gemm(stream, act0, 128, mW[it][1], mB[it][1], act1, 128, NEDGEc, 128, 128, 1);
    launch_gemm(stream, act1, 128, mW[it][2], mB[it][2], worb, 160, NEDGEc, 160, 128, 0);
    // _sum_interaction
    const float* ccbase = con_co + (size_t)it * 4 * 3 * 1024;
    k_contract<<<(N128 + 255) / 256, 256, 0, stream>>>(corb, co, ccbase, spec,
        3072, 0, 0.17677669529f, ave, 0, 0);
    k_iterorb<<<(N128 + 255) / 256, 256, 0, stream>>>(iterb, corb, worb, nl, ave);
    k_edge_reduce<<<(N128 + 255) / 256, 256, 0, stream>>>(sphb, worb, 160,
        64, 96, 96, 96, 1, 0, sph, ave, 1);
    k_contract<<<(N128 + 255) / 256, 256, 0, stream>>>(tmpA, iterb, ccbase, spec,
        3072, 1024, 0.17677669529f, ave, 0, 0);
    k_contract<<<(N128 + 255) / 256, 256, 0, stream>>>(tmpA, sphb, ccbase, spec,
        3072, 2048, 0.17677669529f, ave, 0, 1);
    k_comb<<<(N128 + 255) / 256, 256, 0, stream>>>(co, tmpA, l_co + (size_t)it * 256, spec);
    // radial = ead[it](ead)
    launch_gemm(stream, eadb, 128, eW[it][0], eB[it][0], act0, 128, NEDGEc, 128, eadK, 1);
    launch_gemm(stream, act0, 128, eW[it][1], eB[it][1], act1, 128, NEDGEc, 128, 128, 1);
    launch_gemm(stream, act1, 128, eW[it][2], eB[it][2], radb, 192, NEDGEc, 192, 128, 0);
  }

  // final density block: norm 1/sqrt(6), sum over all 3 radial slices
  k_scalebuf<<<(N128 + 255) / 256, 256, 0, stream>>>(ncorb, co, 0.40824829f, N128);
  k_edge_reduce<<<(N128 + 255) / 256, 256, 0, stream>>>(cotmp, radb, 192,
      0, 32, 32, 32, 3, 64, sph, ave, 0);
  k_density1<<<(N32 + 255) / 256, 256, 0, stream>>>(density, 96, cotmp, ncorb);

  // outnn
  k_densin<<<(N128 + 255) / 256, 256, 0, stream>>>(density, ave, densin);
  launch_gemm(stream, densin, 128, oW0, oB0, hidden, 64, NNODEc, 64, 128, 1);
  k_atom<<<(NNODEc + 255) / 256, 256, 0, stream>>>(hidden, oW1, oB1, scale, spec, cfac, atom);
  k_graph<<<NGRAPHc, 256, 0, stream>>>(atom, out);
  k_total<<<1, 128, 0, stream>>>(out);
}
